// VAE_67224828117441
// MI455X (gfx1250) — matrix operand order repaired and checked by reading, not compiled, not run on hardware
//
#include <hip/hip_runtime.h>
#include <hip/hip_bf16.h>
#include <math.h>

// ---------------------------------------------------------------------------
// CDNA5 (gfx1250) WMMA bf16 implementation of the LSTM seq2seq forward pass.
// B=32, T=128, V=32000, E=512, H=1024.
//
// GEMM core: one wave computes a 32(M) x 64(N) tile with 8 fp32 accumulators
// (v_wmma_f32_16x16x32_bf16), fragments double-buffered across k-steps.
// __launch_bounds__(128, 1) raises the VGPR budget so the ~200-VGPR live set
// (64 acc + 2x48 fragments + addressing) allocates without spills.
// dec_W(bf16)=65MB + h_seq(bf16)=8MB are L2-resident (192MB L2).
// ---------------------------------------------------------------------------

typedef __attribute__((ext_vector_type(16))) __bf16 v16bf;
typedef __attribute__((ext_vector_type(8)))  float  v8f;

union FragU { v16bf v; uint4 q[2]; };

// A-matrix 16x32 bf16 fragment (row-major src, leading dim ld elements).
// lane 0-15: M=lane, K in {0..7,16..23}; lane 16-31: M=lane-16, K in {8..15,24..31}
__device__ __forceinline__ v16bf load_a_frag(const __bf16* __restrict__ base, int ld) {
  int lane = threadIdx.x & 31;
  int r  = lane & 15;
  int kb = (lane >> 4) * 8;
  const __bf16* p = base + r * ld + kb;
  FragU f;
  f.q[0] = *reinterpret_cast<const uint4*>(p);        // K = kb .. kb+7
  f.q[1] = *reinterpret_cast<const uint4*>(p + 16);   // K = kb+16 .. kb+23
  return f.v;
}

// B-matrix 32x16 bf16 fragment. Weight stored (N,K) row-major => "row" = column n.
// lane 0-15: N=lane, K in {0..7,16..23}; lane 16-31: N=lane-16, K in {8..15,24..31}
__device__ __forceinline__ v16bf load_b_frag(const __bf16* __restrict__ base, int ld) {
  int lane = threadIdx.x & 31;
  int n  = lane & 15;
  int kb = (lane >> 4) * 8;
  const __bf16* p = base + n * ld + kb;
  FragU f;
  f.q[0] = *reinterpret_cast<const uint4*>(p);
  f.q[1] = *reinterpret_cast<const uint4*>(p + 16);
  return f.v;
}

struct Frags {
  v16bf a0, a1;   // two M-halves of the 32-row A tile
  v16bf b[4];     // four 16-col B tiles
};

__device__ __forceinline__ void load_frags(Frags& f,
                                           const __bf16* __restrict__ A, int lda,
                                           const __bf16* __restrict__ Bw, int ldb,
                                           int k0) {
  f.a0 = load_a_frag(A + k0, lda);
  f.a1 = load_a_frag(A + 16 * lda + k0, lda);
#pragma unroll
  for (int j = 0; j < 4; ++j)
    f.b[j] = load_b_frag(Bw + j * 16 * ldb + k0, ldb);
}

__device__ __forceinline__ void mma8(const Frags& f, v8f acc[8]) {
#pragma unroll
  for (int j = 0; j < 4; ++j) {
    acc[j] = __builtin_amdgcn_wmma_f32_16x16x32_bf16(
        false, f.a0, false, f.b[j], (short)0, acc[j], false, false);
    acc[4 + j] = __builtin_amdgcn_wmma_f32_16x16x32_bf16(
        false, f.a1, false, f.b[j], (short)0, acc[4 + j], false, false);
  }
}

// 32(M) x 64(N) x K GEMM tile per wave, software-pipelined (K/32 must be even).
__device__ __forceinline__ void gemm32x64(const __bf16* __restrict__ A, int lda,
                                          const __bf16* __restrict__ Bw, int ldb,
                                          int K, v8f acc[8]) {
  const int S = K >> 5;                 // k-steps of 32
  Frags cur, nxt;
  load_frags(cur, A, lda, Bw, ldb, 0);
  for (int s = 1; s < S; s += 2) {
    load_frags(nxt, A, lda, Bw, ldb, s * 32);
    mma8(cur, acc);
    if (s + 1 < S) load_frags(cur, A, lda, Bw, ldb, (s + 1) * 32);
    mma8(nxt, acc);
  }
}

// ---------------------------------------------------------------------------
// Small utility kernels
// ---------------------------------------------------------------------------

__global__ void k_f32_to_bf16(const float* __restrict__ s, __bf16* __restrict__ d, int n) {
  int i = blockIdx.x * blockDim.x + threadIdx.x;
  if (i < n) d[i] = (__bf16)s[i];
}

// x_seq = embed_table[hypothesis[:, :-1]]  -> bf16, shape (32*127, 512)
__global__ void k_gather_x(const float* __restrict__ emb, const int* __restrict__ hyp,
                           __bf16* __restrict__ x) {
  int gid = blockIdx.x * blockDim.x + threadIdx.x;
  if (gid >= 4064 * 512) return;
  int row = gid >> 9;          // b*127 + t
  int e   = gid & 511;
  int b = row / 127, t = row % 127;
  int tok = hyp[b * 128 + t];
  x[gid] = (__bf16)emb[(size_t)tok * 512 + e];
}

// premise embeddings -> bf16, shape (32, 512)
__global__ void k_gather_prem(const float* __restrict__ emb, const int* __restrict__ prem,
                              __bf16* __restrict__ x) {
  int gid = blockIdx.x * blockDim.x + threadIdx.x;
  if (gid >= 32 * 512) return;
  int b = gid >> 9;
  int e = gid & 511;
  int tok = prem[b];
  x[gid] = (__bf16)emb[(size_t)tok * 512 + e];
}

// ---------------------------------------------------------------------------
// GEMM kernels (bf16 WMMA, fp32 accumulate; wave tile 32x64, block = 4 waves)
// ---------------------------------------------------------------------------

// feat = premx(32x512) @ enc_W^T(1024x512) + enc_b  -> c0 (fp32) and h0 (bf16)
// grid (4), block 128
__global__ __launch_bounds__(128, 1)
void k_enc_gemm(const __bf16* __restrict__ X, const __bf16* __restrict__ W,
                const float* __restrict__ bias,
                float* __restrict__ c0, __bf16* __restrict__ hbf) {
  int wave  = threadIdx.x >> 5;
  int nBase = blockIdx.x * 256 + wave * 64;
  v8f acc[8] = {};
  gemm32x64(X, 512, W + nBase * 512, 512, 512, acc);
  int lane = threadIdx.x & 31;
  int col0 = lane & 15;
  int mo   = (lane >> 4) * 8;
#pragma unroll
  for (int half = 0; half < 2; ++half) {
#pragma unroll
    for (int v = 0; v < 8; ++v) {
      int m = half * 16 + mo + v;
#pragma unroll
      for (int j = 0; j < 4; ++j) {
        int col = nBase + j * 16 + col0;
        float val = acc[half * 4 + j][v] + bias[col];
        c0[m * 1024 + col]  = val;
        hbf[m * 1024 + col] = (__bf16)val;
      }
    }
  }
}

// xg = x(4064x512) @ W_ih^T(4096x512) + (b_ih + b_hh)  -> fp32 (4064x4096)
// grid (16, 127), block 128
__global__ __launch_bounds__(128, 1)
void k_xg_gemm(const __bf16* __restrict__ X, const __bf16* __restrict__ Wih,
               const float* __restrict__ b_ih, const float* __restrict__ b_hh,
               float* __restrict__ xg) {
  int wave  = threadIdx.x >> 5;
  int mBase = blockIdx.y * 32;
  int nBase = blockIdx.x * 256 + wave * 64;
  v8f acc[8] = {};
  gemm32x64(X + mBase * 512, 512, Wih + nBase * 512, 512, 512, acc);
  int lane = threadIdx.x & 31;
  int col0 = lane & 15;
  int mo   = (lane >> 4) * 8;
#pragma unroll
  for (int half = 0; half < 2; ++half) {
#pragma unroll
    for (int v = 0; v < 8; ++v) {
      int m = mBase + half * 16 + mo + v;
#pragma unroll
      for (int j = 0; j < 4; ++j) {
        int col = nBase + j * 16 + col0;
        xg[m * 4096 + col] = acc[half * 4 + j][v] + b_ih[col] + b_hh[col];
      }
    }
  }
}

// gates = h(32x1024,bf16) @ W_hh^T(4096x1024) + xg[:, t, :]  -> fp32 (32x4096)
// grid (16), block 128; each W_hh fragment reused for both M-halves.
__global__ __launch_bounds__(128, 1)
void k_step_gemm(const __bf16* __restrict__ h, const __bf16* __restrict__ Whh,
                 const float* __restrict__ xg, float* __restrict__ gates, int t) {
  int wave  = threadIdx.x >> 5;
  int nBase = blockIdx.x * 256 + wave * 64;
  v8f acc[8] = {};
  gemm32x64(h, 1024, Whh + nBase * 1024, 1024, 1024, acc);
  int lane = threadIdx.x & 31;
  int col0 = lane & 15;
  int mo   = (lane >> 4) * 8;
#pragma unroll
  for (int half = 0; half < 2; ++half) {
#pragma unroll
    for (int v = 0; v < 8; ++v) {
      int m = half * 16 + mo + v;          // batch index 0..31
      int xrow = (m * 127 + t) * 4096;
#pragma unroll
      for (int j = 0; j < 4; ++j) {
        int col = nBase + j * 16 + col0;
        gates[m * 4096 + col] = acc[half * 4 + j][v] + xg[xrow + col];
      }
    }
  }
}

// LSTM elementwise: c,h update; emit h in bf16 (state + h_seq[t])
__global__ void k_step_elem(const float* __restrict__ gates, float* __restrict__ c,
                            __bf16* __restrict__ hbf, __bf16* __restrict__ hseq, int t) {
  int gid = blockIdx.x * blockDim.x + threadIdx.x;
  if (gid >= 32 * 1024) return;
  int b = gid >> 10, n = gid & 1023;
  float ig = gates[b * 4096 + n];
  float fg = gates[b * 4096 + 1024 + n];
  float gg = gates[b * 4096 + 2048 + n];
  float og = gates[b * 4096 + 3072 + n];
  ig = 1.0f / (1.0f + expf(-ig));
  fg = 1.0f / (1.0f + expf(-fg));
  gg = tanhf(gg);
  og = 1.0f / (1.0f + expf(-og));
  float cn = fg * c[gid] + ig * gg;
  float hn = og * tanhf(cn);
  c[gid] = cn;
  hbf[gid] = (__bf16)hn;
  hseq[(b * 127 + t) * 1024 + n] = (__bf16)hn;
}

// logits = h_seq(4064x1024) @ dec_W^T(32000x1024) + dec_b, stored time-shifted
// into raw_outputs region of d_out. grid (125, 127), block 128.
__global__ __launch_bounds__(128, 1)
void k_dec_gemm(const __bf16* __restrict__ Hs, const __bf16* __restrict__ Wd,
                const float* __restrict__ db, float* __restrict__ out) {
  int wave  = threadIdx.x >> 5;
  int mBase = blockIdx.y * 32;
  int nBase = blockIdx.x * 256 + wave * 64;
  v8f acc[8] = {};
  gemm32x64(Hs + mBase * 1024, 1024, Wd + (size_t)nBase * 1024, 1024, 1024, acc);
  int lane = threadIdx.x & 31;
  int col0 = lane & 15;
  int mo   = (lane >> 4) * 8;
#pragma unroll
  for (int half = 0; half < 2; ++half) {
#pragma unroll
    for (int v = 0; v < 8; ++v) {
      int r  = mBase + half * 16 + mo + v;   // 0..4063 = b*127 + (t-1)
      int b  = r / 127;
      int tt = r % 127;
      float* orow = out + 4096 + (size_t)(b * 128 + tt + 1) * 32000;
#pragma unroll
      for (int j = 0; j < 4; ++j) {
        int col = nBase + j * 16 + col0;
        orow[col] = acc[half * 4 + j][v] + db[col];
      }
    }
  }
}

// raw_outputs[:, 0, :] = 0 and outputted_words[:, 0] = hypothesis[:, 0]
__global__ void k_init_out(float* __restrict__ out, const int* __restrict__ hyp) {
  int gid = blockIdx.x * blockDim.x + threadIdx.x;
  if (gid < 32 * 32000) {
    int b = gid / 32000, v = gid % 32000;
    out[4096 + (size_t)(b * 128) * 32000 + v] = 0.0f;
  }
  if (gid < 32) out[gid * 128] = (float)hyp[gid * 128];
}

// outputted_words[b, t+1] = argmax_v logits[b, t, v]  (first-max tie break)
__global__ void k_argmax(const float* __restrict__ out, float* __restrict__ words) {
  __shared__ float smax[256];
  __shared__ int   sidx[256];
  int r = blockIdx.x;                 // 0..4063
  int b = r / 127, tt = r % 127;
  const float* row = out + 4096 + (size_t)(b * 128 + tt + 1) * 32000;
  float best = -INFINITY;
  int bi = 0x7fffffff;
  for (int v = threadIdx.x; v < 32000; v += 256) {
    float x = row[v];
    if (x > best || (x == best && v < bi)) { best = x; bi = v; }
  }
  smax[threadIdx.x] = best;
  sidx[threadIdx.x] = bi;
  __syncthreads();
  for (int s = 128; s > 0; s >>= 1) {
    if (threadIdx.x < s) {
      float ox = smax[threadIdx.x + s];
      int   oi = sidx[threadIdx.x + s];
      if (ox > smax[threadIdx.x] || (ox == smax[threadIdx.x] && oi < sidx[threadIdx.x])) {
        smax[threadIdx.x] = ox;
        sidx[threadIdx.x] = oi;
      }
    }
    __syncthreads();
  }
  if (threadIdx.x == 0) words[b * 128 + tt + 1] = (float)sidx[0];
}

// ---------------------------------------------------------------------------
// Host launcher
// ---------------------------------------------------------------------------

extern "C" void kernel_launch(void* const* d_in, const int* in_sizes, int n_in,
                              void* d_out, int out_size, void* d_ws, size_t ws_size,
                              hipStream_t stream) {
  (void)in_sizes; (void)n_in; (void)out_size; (void)ws_size;
  const int*   premises = (const int*)d_in[0];
  const int*   hyp      = (const int*)d_in[1];
  // d_in[2] = labels (unused)
  const float* emb      = (const float*)d_in[3];
  const float* enc_W    = (const float*)d_in[4];
  const float* enc_b    = (const float*)d_in[5];
  const float* W_ih     = (const float*)d_in[6];
  const float* W_hh     = (const float*)d_in[7];
  const float* b_ih     = (const float*)d_in[8];
  const float* b_hh     = (const float*)d_in[9];
  const float* dec_W    = (const float*)d_in[10];
  const float* dec_b    = (const float*)d_in[11];
  float* out = (float*)d_out;

  // Workspace layout (all offsets 16-byte aligned)
  char* w = (char*)d_ws;
  __bf16* encW_bf = (__bf16*)w;  w += (size_t)1024 * 512 * 2;
  __bf16* Wih_bf  = (__bf16*)w;  w += (size_t)4096 * 512 * 2;
  __bf16* Whh_bf  = (__bf16*)w;  w += (size_t)4096 * 1024 * 2;
  __bf16* decW_bf = (__bf16*)w;  w += (size_t)32000 * 1024 * 2;
  __bf16* premx   = (__bf16*)w;  w += (size_t)32 * 512 * 2;
  __bf16* x_bf    = (__bf16*)w;  w += (size_t)4064 * 512 * 2;
  __bf16* h_bf    = (__bf16*)w;  w += (size_t)32 * 1024 * 2;
  __bf16* hseq    = (__bf16*)w;  w += (size_t)4064 * 1024 * 2;
  float*  xg      = (float*)w;   w += (size_t)4064 * 4096 * 4;
  float*  gates   = (float*)w;   w += (size_t)32 * 4096 * 4;
  float*  c_st    = (float*)w;   w += (size_t)32 * 1024 * 4;

  // 1) weight conversions fp32 -> bf16
  k_f32_to_bf16<<<(1024 * 512 + 255) / 256, 256, 0, stream>>>(enc_W, encW_bf, 1024 * 512);
  k_f32_to_bf16<<<(4096 * 512 + 255) / 256, 256, 0, stream>>>(W_ih, Wih_bf, 4096 * 512);
  k_f32_to_bf16<<<(4096 * 1024 + 255) / 256, 256, 0, stream>>>(W_hh, Whh_bf, 4096 * 1024);
  k_f32_to_bf16<<<(32000 * 1024 + 255) / 256, 256, 0, stream>>>(dec_W, decW_bf, 32000 * 1024);

  // 2) embedding gathers
  k_gather_prem<<<(32 * 512 + 255) / 256, 256, 0, stream>>>(emb, premises, premx);
  k_gather_x<<<(4064 * 512 + 255) / 256, 256, 0, stream>>>(emb, hyp, x_bf);

  // 3) encoder: h0 = c0 = premx @ enc_W^T + enc_b
  k_enc_gemm<<<4, 128, 0, stream>>>(premx, encW_bf, enc_b, c_st, h_bf);

  // 4) xg = x @ W_ih^T + (b_ih + b_hh)  (batched over all timesteps)
  k_xg_gemm<<<dim3(16, 127), 128, 0, stream>>>(x_bf, Wih_bf, b_ih, b_hh, xg);

  // 5) LSTM recurrence (sequential over 127 steps)
  for (int t = 0; t < 127; ++t) {
    k_step_gemm<<<16, 128, 0, stream>>>(h_bf, Whh_bf, xg, gates, t);
    k_step_elem<<<(32 * 1024 + 255) / 256, 256, 0, stream>>>(gates, c_st, h_bf, hseq, t);
  }

  // 6) output zero row + first outputted word
  k_init_out<<<(32 * 32000 + 255) / 256, 256, 0, stream>>>(out, hyp);

  // 7) decoder GEMM -> raw_outputs (time-shifted)
  k_dec_gemm<<<dim3(125, 127), 128, 0, stream>>>(hseq, decW_bf, dec_b, out);

  // 8) argmax -> outputted_words[:, 1:]
  k_argmax<<<4064, 256, 0, stream>>>(out, out);
}
